// GATConvolution_72911364817010
// MI455X (gfx1250) — matrix operand order repaired and checked by reading, not compiled, not run on hardware
//
#include <hip/hip_runtime.h>
#include <hip/hip_bf16.h>

// ---------------------------------------------------------------------------
// GAT (2-layer, single head) for MI455X / gfx1250.
// Dense GEMMs: v_wmma_f32_16x16x32_f16 (wave32 WMMA), weights staged in LDS
// as f16 column-major (ds_load_b128 fragments), K-loop fully unrolled.
// Edge softmax + scatter-aggregation: L2-resident atomic passes
// (h = 51MB fits the 192MB L2, so random gathers/atomics stay on-die).
// ---------------------------------------------------------------------------

#define GAT_N    100000
#define GAT_E    1600000
#define GAT_FIN  256
#define GAT_HID  128
#define GAT_NCLS 64
#define NEG_SLOPE 0.2f

typedef __attribute__((ext_vector_type(16))) _Float16 v16h;
typedef __attribute__((ext_vector_type(8)))  float    v8f;

// ---------------------------------------------------------------------------
// WMMA GEMM: C[M x Nc] = A[M x K] * B[K x Nc], row-major fp32 in/out.
// Operands converted to f16; f32 accumulate. M%16==0, K%32==0, Nc%16==0.
// One wave per 16x16 C tile. blockDim.x = 32*(Nc/16) = 2*Nc, grid.x = M/16.
// B is staged once per block into LDS, transposed (sB[n*K + k]) so each
// lane's fragment is 16 contiguous halves (32B-aligned -> ds_load_b128 x2).
// ---------------------------------------------------------------------------
template <int K, int Nc>
__global__ void gat_gemm_wmma(const float* __restrict__ A,
                              const float* __restrict__ B,
                              float* __restrict__ C) {
  __shared__ _Float16 sB[K * Nc];   // <= 64KB for K=256,Nc=128 (320KB/WGP)

  // Cooperative transposing convert of W into LDS (coalesced global reads:
  // consecutive tid -> consecutive n within a row of W).
  for (int idx = threadIdx.x; idx < K * Nc; idx += 2 * Nc) {
    const int k  = idx / Nc;
    const int n2 = idx % Nc;
    sB[n2 * K + k] = (_Float16)B[(size_t)k * Nc + n2];
  }
  __syncthreads();

  const int lane    = threadIdx.x & 31;
  const int wave    = threadIdx.x >> 5;
  const int rowTile = blockIdx.x << 4;
  const int colTile = wave << 4;

  const int halfSel = lane >> 4;           // 0 = lanes 0-15, 1 = lanes 16-31
  const int m = rowTile + (lane & 15);     // A row held by this lane
  const int n = colTile + (lane & 15);     // B col held by this lane

  const float* __restrict__ arow = A + (size_t)m * K;
  const _Float16* __restrict__ bcol = sB + n * K;

  v8f acc = {};
#pragma unroll
  for (int k0 = 0; k0 < K; k0 += 32) {
    // A fragment (contiguous k order, same as B):
    //  lanes 0-15 : V0-7 -> K=k0+0..15
    //  lanes16-31 : V0-7 -> K=k0+16..31
    const int ka0 = k0 + halfSel * 16;       // 32B aligned
    const int ka1 = ka0 + 8;                 // 32B aligned
    const float4 x0 = *reinterpret_cast<const float4*>(arow + ka0);
    const float4 x1 = *reinterpret_cast<const float4*>(arow + ka0 + 4);
    const float4 x2 = *reinterpret_cast<const float4*>(arow + ka1);
    const float4 x3 = *reinterpret_cast<const float4*>(arow + ka1 + 4);
    v16h a;
    a[0]  = (_Float16)x0.x; a[1]  = (_Float16)x0.y; a[2]  = (_Float16)x0.z; a[3]  = (_Float16)x0.w;
    a[4]  = (_Float16)x1.x; a[5]  = (_Float16)x1.y; a[6]  = (_Float16)x1.z; a[7]  = (_Float16)x1.w;
    a[8]  = (_Float16)x2.x; a[9]  = (_Float16)x2.y; a[10] = (_Float16)x2.z; a[11] = (_Float16)x2.w;
    a[12] = (_Float16)x3.x; a[13] = (_Float16)x3.y; a[14] = (_Float16)x3.z; a[15] = (_Float16)x3.w;

    // B fragment (contiguous k order, same as A): lane holds col n;
    //  lanes 0-15 : K=k0+0..15 ; lanes 16-31 : K=k0+16..31.
    // 16 contiguous halves in LDS, 32B aligned.
    const v16h b = *reinterpret_cast<const v16h*>(bcol + k0 + halfSel * 16);

    acc = __builtin_amdgcn_wmma_f32_16x16x32_f16(false, a, false, b,
                                                 (short)0, acc, false, false);
  }

  // C/D layout: VGPR r -> (M = rowTile + r + 8*halfSel, N = n)
  float* __restrict__ crow = C + (size_t)(rowTile + halfSel * 8) * Nc + n;
#pragma unroll
  for (int r = 0; r < 8; ++r) crow[(size_t)r * Nc] = acc[r];
}

// ---------------------------------------------------------------------------
// Per-node attention scores: s_src[i] = h[i,:]@a_src, s_dst[i] = h[i,:]@a_dst
// ---------------------------------------------------------------------------
__global__ void gat_node_scores(const float* __restrict__ h,
                                const float* __restrict__ a_src,
                                const float* __restrict__ a_dst,
                                float* __restrict__ s_src,
                                float* __restrict__ s_dst,
                                int n_nodes, int F) {
  int i = blockIdx.x * blockDim.x + threadIdx.x;
  if (i >= n_nodes) return;
  const float* __restrict__ row = h + (size_t)i * F;
  float ss = 0.f, sd = 0.f;
  for (int f = 0; f < F; f += 4) {
    float4 v = *reinterpret_cast<const float4*>(row + f);
    ss += v.x * a_src[f + 0] + v.y * a_src[f + 1] + v.z * a_src[f + 2] + v.w * a_src[f + 3];
    sd += v.x * a_dst[f + 0] + v.y * a_dst[f + 1] + v.z * a_dst[f + 2] + v.w * a_dst[f + 3];
  }
  s_src[i] = ss;
  s_dst[i] = sd;
}

__global__ void gat_fill_f32(float* __restrict__ p, float v, size_t n) {
  size_t i = (size_t)blockIdx.x * blockDim.x + threadIdx.x;
  if (i < n) p[i] = v;
}

__device__ __forceinline__ void atomicMaxFloat(float* addr, float value) {
  // IEEE ordering trick; addr initialized to -inf.
  if (value >= 0.0f)
    atomicMax(reinterpret_cast<int*>(addr), __float_as_int(value));
  else
    atomicMin(reinterpret_cast<unsigned int*>(addr), __float_as_uint(value));
}

__device__ __forceinline__ void edge_endpoints(int e, const int* __restrict__ esrc,
                                               const int* __restrict__ edst,
                                               int& s, int& d) {
  if (e < GAT_E) { s = esrc[e]; d = edst[e]; }
  else           { s = e - GAT_E; d = e - GAT_E; }   // self loops
}

// Pass 1: per-edge leaky-relu logit; store it; segment max over dst.
__global__ void gat_edge_logit_max(const int* __restrict__ esrc,
                                   const int* __restrict__ edst,
                                   const float* __restrict__ s_src,
                                   const float* __restrict__ s_dst,
                                   float* __restrict__ elog,
                                   float* __restrict__ mx) {
  int e = blockIdx.x * blockDim.x + threadIdx.x;
  if (e >= GAT_E + GAT_N) return;
  int s, d; edge_endpoints(e, esrc, edst, s, d);
  float z = s_src[s] + s_dst[d];
  z = (z > 0.f) ? z : NEG_SLOPE * z;
  elog[e] = z;
  atomicMaxFloat(mx + d, z);
}

// Pass 2: e = exp(logit - m[dst]) stored in place; segment sum into denom.
__global__ void gat_edge_exp_sum(const int* __restrict__ esrc,
                                 const int* __restrict__ edst,
                                 float* __restrict__ elog,
                                 const float* __restrict__ mx,
                                 float* __restrict__ denom) {
  int e = blockIdx.x * blockDim.x + threadIdx.x;
  if (e >= GAT_E + GAT_N) return;
  int s, d; edge_endpoints(e, esrc, edst, s, d);
  float ev = __expf(elog[e] - mx[d]);
  elog[e] = ev;
  atomicAdd(denom + d, ev);
}

// Pass 3: agg[dst,:] += (e/denom[dst]) * h[src,:].  One wave per edge.
__global__ void gat_edge_aggregate(const int* __restrict__ esrc,
                                   const int* __restrict__ edst,
                                   const float* __restrict__ elog,
                                   const float* __restrict__ denom,
                                   const float* __restrict__ h,
                                   float* __restrict__ agg,
                                   int F) {
  const int wavesPerBlock = blockDim.x >> 5;
  int e = blockIdx.x * wavesPerBlock + (threadIdx.x >> 5);
  if (e >= GAT_E + GAT_N) return;
  const int lane = threadIdx.x & 31;
  int s, d; edge_endpoints(e, esrc, edst, s, d);
  float alpha = elog[e] / denom[d];
  const float* __restrict__ hs = h + (size_t)s * F;
  float* __restrict__ od = agg + (size_t)d * F;
  for (int f = lane * 4; f < F; f += 128) {
    float4 v = *reinterpret_cast<const float4*>(hs + f);
    atomicAdd(od + f + 0, alpha * v.x);
    atomicAdd(od + f + 1, alpha * v.y);
    atomicAdd(od + f + 2, alpha * v.z);
    atomicAdd(od + f + 3, alpha * v.w);
  }
}

// out = agg + bias (optionally ReLU).
__global__ void gat_finalize(const float* __restrict__ agg,
                             const float* __restrict__ bias,
                             float* __restrict__ outp,
                             int F, int doRelu, size_t total) {
  size_t i = (size_t)blockIdx.x * blockDim.x + threadIdx.x;
  if (i >= total) return;
  float v = agg[i] + bias[i % F];
  if (doRelu) v = fmaxf(v, 0.f);
  outp[i] = v;
}

// ---------------------------------------------------------------------------
extern "C" void kernel_launch(void* const* d_in, const int* in_sizes, int n_in,
                              void* d_out, int out_size, void* d_ws, size_t ws_size,
                              hipStream_t stream) {
  const float* x      = (const float*)d_in[0];
  const int*   ei     = (const int*)  d_in[1];
  const float* W1     = (const float*)d_in[2];
  const float* a1_src = (const float*)d_in[3];
  const float* a1_dst = (const float*)d_in[4];
  const float* b1     = (const float*)d_in[5];
  const float* W2     = (const float*)d_in[6];
  const float* a2_src = (const float*)d_in[7];
  const float* a2_dst = (const float*)d_in[8];
  const float* b2     = (const float*)d_in[9];
  float* out = (float*)d_out;

  const int* esrc = ei;
  const int* edst = ei + GAT_E;

  const int Etot = GAT_E + GAT_N;

  // ---- workspace carve-up (all fp32) --------------------------------------
  char* ws = (char*)d_ws;
  auto take = [&](size_t nElem) { float* p = (float*)ws; ws += ((nElem * 4 + 255) & ~(size_t)255); return p; };
  float* bufA  = take((size_t)GAT_N * GAT_HID);  // h1, then relu(h1_agg+b1) = layer2 input
  float* bufB  = take((size_t)GAT_N * GAT_HID);  // agg1; then [h2 | agg2] halves
  float* sSrc  = take(GAT_N);
  float* sDst  = take(GAT_N);
  float* mx    = take(GAT_N);
  float* denom = take(GAT_N);
  float* elog  = take(Etot);
  float* h2    = bufB;                           // N x 64 (agg1 consumed by then)
  float* agg2  = bufB + (size_t)GAT_N * GAT_NCLS;

  const float negInf = -__builtin_huge_valf();
  auto gridFor = [](size_t n, int bs) { return (unsigned)((n + bs - 1) / bs); };

  // =================== Layer 1: x[N,256] -> relu(GAT) [N,128] ==============
  gat_gemm_wmma<GAT_FIN, GAT_HID><<<GAT_N / 16, 2 * GAT_HID, 0, stream>>>(x, W1, bufA);
  gat_node_scores<<<gridFor(GAT_N, 256), 256, 0, stream>>>(bufA, a1_src, a1_dst, sSrc, sDst, GAT_N, GAT_HID);
  gat_fill_f32<<<gridFor(GAT_N, 256), 256, 0, stream>>>(mx, negInf, GAT_N);
  gat_fill_f32<<<gridFor(GAT_N, 256), 256, 0, stream>>>(denom, 0.f, GAT_N);
  gat_fill_f32<<<gridFor((size_t)GAT_N * GAT_HID, 256), 256, 0, stream>>>(bufB, 0.f, (size_t)GAT_N * GAT_HID);
  gat_edge_logit_max<<<gridFor(Etot, 256), 256, 0, stream>>>(esrc, edst, sSrc, sDst, elog, mx);
  gat_edge_exp_sum<<<gridFor(Etot, 256), 256, 0, stream>>>(esrc, edst, elog, mx, denom);
  gat_edge_aggregate<<<gridFor(Etot, 8), 256, 0, stream>>>(esrc, edst, elog, denom, bufA, bufB, GAT_HID);
  gat_finalize<<<gridFor((size_t)GAT_N * GAT_HID, 256), 256, 0, stream>>>(bufB, b1, bufA, GAT_HID, 1, (size_t)GAT_N * GAT_HID);

  // =================== Layer 2: h[N,128] -> GAT [N,64] =====================
  gat_gemm_wmma<GAT_HID, GAT_NCLS><<<GAT_N / 16, 2 * GAT_NCLS, 0, stream>>>(bufA, W2, h2);
  gat_node_scores<<<gridFor(GAT_N, 256), 256, 0, stream>>>(h2, a2_src, a2_dst, sSrc, sDst, GAT_N, GAT_NCLS);
  gat_fill_f32<<<gridFor(GAT_N, 256), 256, 0, stream>>>(mx, negInf, GAT_N);
  gat_fill_f32<<<gridFor(GAT_N, 256), 256, 0, stream>>>(denom, 0.f, GAT_N);
  gat_fill_f32<<<gridFor((size_t)GAT_N * GAT_NCLS, 256), 256, 0, stream>>>(agg2, 0.f, (size_t)GAT_N * GAT_NCLS);
  gat_edge_logit_max<<<gridFor(Etot, 256), 256, 0, stream>>>(esrc, edst, sSrc, sDst, elog, mx);
  gat_edge_exp_sum<<<gridFor(Etot, 256), 256, 0, stream>>>(esrc, edst, elog, mx, denom);
  gat_edge_aggregate<<<gridFor(Etot, 8), 256, 0, stream>>>(esrc, edst, elog, denom, h2, agg2, GAT_NCLS);
  gat_finalize<<<gridFor((size_t)GAT_N * GAT_NCLS, 256), 256, 0, stream>>>(agg2, b2, out, GAT_NCLS, 0, (size_t)GAT_N * GAT_NCLS);
}
